// JointAttentionLayer_59682865545605
// MI455X (gfx1250) — matrix operand order repaired and checked by reading, not compiled, not run on hardware
//
#include <hip/hip_runtime.h>
#include <hip/hip_bf16.h>

typedef __attribute__((ext_vector_type(16))) _Float16 v16h;
typedef __attribute__((ext_vector_type(8)))  _Float16 v8h;
typedef __attribute__((ext_vector_type(8)))  float    v8f;

#define AB_ 8
#define AC_ 512
#define AN_ 2048
#define AC4_ 128

// LDS tile strides (halves): 32 data + 8 pad -> 80B rows, conflict-free frags
#define LDA_S 40
#define LDB_S 40

// ---------------------------------------------------------------------------
// WMMA fragment helpers (CDNA5 16-bit layouts, ISA 7.12.2). Work from both
// global and LDS pointers (16B-aligned v8h loads -> b128s).
// ---------------------------------------------------------------------------
__device__ __forceinline__ v16h load_a_frag(const _Float16* A, int lda) {
  const int lane = threadIdx.x & 31;
  const int row  = lane & 15;
  const int kb   = (lane >> 4) << 3;
  const _Float16* p = A + (size_t)row * lda + kb;
  v8h lo = *(const v8h*)(p);
  v8h hi = *(const v8h*)(p + 16);
  v16h a;
#pragma unroll
  for (int i = 0; i < 8; ++i) { a[i] = lo[i]; a[8 + i] = hi[i]; }
  return a;
}

__device__ __forceinline__ v16h load_b_frag(const _Float16* Bt, int ldb) {
  const int lane = threadIdx.x & 31;
  const int col  = lane & 15;
  const int kb   = (lane >> 4) << 3;
  const _Float16* p = Bt + (size_t)col * ldb + kb;
  v8h lo = *(const v8h*)(p);
  v8h hi = *(const v8h*)(p + 16);
  v16h b;
#pragma unroll
  for (int i = 0; i < 8; ++i) { b[i] = lo[i]; b[8 + i] = hi[i]; }
  return b;
}

__device__ __forceinline__ v8f wmma_f16(v16h a, v16h b, v8f c) {
  return __builtin_amdgcn_wmma_f32_16x16x32_f16(false, a, false, b, (short)0, c, false, false);
}

// ---------------------------------------------------------------------------
// CDNA5 async global->LDS copy (ASYNCcnt-tracked, no VGPR round trip).
// VDST VGPR carries the LDS byte offset (low 32 bits of the flat address of a
// __shared__ object are the wave-relative LDS offset).
// ---------------------------------------------------------------------------
__device__ __forceinline__ void async_copy_b128(uint32_t lds_off, const _Float16* gptr) {
  asm volatile("global_load_async_to_lds_b128 %0, %1, off"
               :: "v"(lds_off), "v"(gptr) : "memory");
}
__device__ __forceinline__ void wait_asynccnt0() {
  asm volatile("s_wait_asynccnt 0x0" ::: "memory");
}
__device__ __forceinline__ uint32_t lds_addr(const void* p) {
  return (uint32_t)(uintptr_t)p;
}

// ---------------------------------------------------------------------------
// Elementwise f32 -> f16 (weights)
// ---------------------------------------------------------------------------
__global__ void k_f32_to_f16(const float* __restrict__ src, _Float16* __restrict__ dst, int n) {
  int i = blockIdx.x * blockDim.x + threadIdx.x;
  if (i < n) dst[i] = (_Float16)src[i];
}

// ---------------------------------------------------------------------------
// Tiled transpose + convert:  X f32 [B][C][N]  ->  XT f16 [B][N][C]
// ---------------------------------------------------------------------------
__global__ void k_transpose_f16(const float* __restrict__ X, _Float16* __restrict__ XT,
                                int C, int N) {
  __shared__ float tile[32][33];
  const int b  = blockIdx.z;
  const int c0 = blockIdx.y * 32;
  const int n0 = blockIdx.x * 32;
  const int tx = threadIdx.x, ty = threadIdx.y;  // (32, 8)
  const float* Xb = X + (size_t)b * C * N;
#pragma unroll
  for (int i = ty; i < 32; i += 8)
    tile[i][tx] = Xb[(size_t)(c0 + i) * N + n0 + tx];
  __syncthreads();
  _Float16* XTb = XT + (size_t)b * N * C;
#pragma unroll
  for (int i = ty; i < 32; i += 8)
    XTb[(size_t)(n0 + i) * C + c0 + tx] = (_Float16)tile[tx][i];
}

// ---------------------------------------------------------------------------
// Blocked NT GEMM:  D[b][m][n] = sum_k A[b][m][k] * Bt[b][n][k]
// Block tile 64(M) x WN*64(N) x 32(K); 8 waves in a 2x4 grid, each wave owns a
// 32 x WN*16 register tile (2*WN wmma accumulators). Operands double-buffered
// in LDS via global_load_async_to_lds_b128; per-wave s_wait_asynccnt + one
// workgroup barrier per K-step (loads for k+1 in flight during wmma on k).
// Epilogues: 0 = f16 store, 1 = +bias f16, 2 = (x - acc) transposed f16,
//            3 = BN(eval)+ReLU+residual f32.
// ---------------------------------------------------------------------------
template <int EPI, int WN>
__global__ __launch_bounds__(256) void k_gemm_big(
    const _Float16* __restrict__ A, size_t sA, int lda,
    const _Float16* __restrict__ Bt, size_t sB, int ldb,
    _Float16* __restrict__ Dh, size_t sD, int lddh,
    const float* __restrict__ bias,
    const _Float16* __restrict__ XT, size_t sXT,
    const float* __restrict__ xin,
    const float* __restrict__ btv, const float* __restrict__ gamma,
    const float* __restrict__ beta, const float* __restrict__ rmean,
    const float* __restrict__ rvar, float* __restrict__ outf,
    int MtB, int NtB, int K, int N_, int C_) {
  constexpr int BN = WN * 64;                // block N tile
  __shared__ _Float16 As[2][64 * LDA_S];
  __shared__ _Float16 Bs[2][BN * LDB_S];

  const int tid  = threadIdx.x;
  const int wv   = tid >> 5;
  const int lane = tid & 31;
  const int wm   = wv >> 2;   // 0..1
  const int wn   = wv & 3;    // 0..3

  const int perBatch = MtB * NtB;
  const int b   = blockIdx.x / perBatch;
  const int rem = blockIdx.x - b * perBatch;
  const int bm  = rem / NtB;
  const int bn  = rem - bm * NtB;

  const _Float16* Ag = A  + (size_t)b * sA + (size_t)(bm * 64) * lda;
  const _Float16* Bg = Bt + (size_t)b * sB + (size_t)(bn * BN) * ldb;

  // staging assignments (16B chunks); A: 256 chunks, B: BN*4 chunks
  const int arow = tid >> 2, ach = (tid & 3) << 3;

  const int nk = K >> 5;
  // prologue: stage k-step 0 into buffer 0
  {
    async_copy_b128(lds_addr(&As[0][arow * LDA_S + ach]), Ag + (size_t)arow * lda + ach);
#pragma unroll
    for (int j = 0; j < WN; ++j) {
      int q = tid + 256 * j;
      int row = q >> 2, ch = (q & 3) << 3;
      async_copy_b128(lds_addr(&Bs[0][row * LDB_S + ch]), Bg + (size_t)row * ldb + ch);
    }
  }

  v8f acc[2][WN] = {};
  for (int kk = 0; kk < nk; ++kk) {
    wait_asynccnt0();
    __syncthreads();                          // buffer kk&1 ready for everyone
    if (kk + 1 < nk) {
      const int nb = (kk + 1) & 1;
      const int k0 = (kk + 1) << 5;
      async_copy_b128(lds_addr(&As[nb][arow * LDA_S + ach]),
                      Ag + (size_t)arow * lda + k0 + ach);
#pragma unroll
      for (int j = 0; j < WN; ++j) {
        int q = tid + 256 * j;
        int row = q >> 2, ch = (q & 3) << 3;
        async_copy_b128(lds_addr(&Bs[nb][row * LDB_S + ch]),
                        Bg + (size_t)row * ldb + k0 + ch);
      }
    }
    const _Float16* Asp = &As[kk & 1][0];
    const _Float16* Bsp = &Bs[kk & 1][0];
    v16h af[2];
#pragma unroll
    for (int mi = 0; mi < 2; ++mi)
      af[mi] = load_a_frag(Asp + (wm * 32 + mi * 16) * LDA_S, LDA_S);
    v16h bf[WN];
#pragma unroll
    for (int ni = 0; ni < WN; ++ni)
      bf[ni] = load_b_frag(Bsp + (wn * WN * 16 + ni * 16) * LDB_S, LDB_S);
#pragma unroll
    for (int mi = 0; mi < 2; ++mi)
#pragma unroll
      for (int ni = 0; ni < WN; ++ni)
        acc[mi][ni] = wmma_f16(af[mi], bf[ni], acc[mi][ni]);
  }

  // epilogue
  const int col = lane & 15;
  const int r0  = (lane >> 4) << 3;
#pragma unroll
  for (int mi = 0; mi < 2; ++mi) {
#pragma unroll
    for (int ni = 0; ni < WN; ++ni) {
      const int Rb = bm * 64 + wm * 32 + mi * 16 + r0;
      const int Cb = bn * BN + wn * WN * 16 + ni * 16 + col;
      v8f a_ = acc[mi][ni];
      if constexpr (EPI == 0) {
        _Float16* Dp = Dh + (size_t)b * sD + (size_t)Rb * lddh + Cb;
#pragma unroll
        for (int v = 0; v < 8; ++v) Dp[(size_t)v * lddh] = (_Float16)a_[v];
      } else if constexpr (EPI == 1) {
        _Float16* Dp = Dh + (size_t)b * sD + (size_t)Rb * lddh + Cb;
#pragma unroll
        for (int v = 0; v < 8; ++v) Dp[(size_t)v * lddh] = (_Float16)(a_[v] + bias[Rb + v]);
      } else if constexpr (EPI == 2) {
        const size_t off = (size_t)Cb * C_ + Rb;
        v8h xt = *(const v8h*)(XT + (size_t)b * sXT + off);
        v8h h;
#pragma unroll
        for (int v = 0; v < 8; ++v) h[v] = (_Float16)((float)xt[v] - a_[v]);
        *(v8h*)(Dh + (size_t)b * sD + off) = h;
      } else {
#pragma unroll
        for (int v = 0; v < 8; ++v) {
          const int c = Rb + v;
          const float alpha = gamma[c] * rsqrtf(rvar[c] + 1e-5f);
          const float bb    = (btv[c] - rmean[c]) * alpha + beta[c];
          const size_t idx  = ((size_t)b * C_ + c) * N_ + Cb;
          outf[idx] = xin[idx] + fmaxf(a_[v] * alpha + bb, 0.0f);
        }
      }
    }
  }
}

// ---------------------------------------------------------------------------
// Attention: per block -> batch b, 16 query rows; WMMA energy into LDS,
// reference-exact masked column scaling (d[j] = mask[j]*cnt + 1e-9), rowwise
// softmax, attention written TRANSPOSED f16 [B][m][n] (K-contiguous B operand
// for the x_r GEMM).
// ---------------------------------------------------------------------------
__device__ __forceinline__ float att_xform(float e, int mr, int mc, float invcnt) {
  if (mr && mc) return e * invcnt;
  return mc ? -1.0e4f : -3.0e4f;   // sentinels preserve fully-masked-row case
}

__global__ __launch_bounds__(256) void k_attention(
    const _Float16* __restrict__ XQ,   // [B][N][C4]
    const _Float16* __restrict__ XKT,  // [B][N][C4]
    const int* __restrict__ mask,      // [B][N]
    _Float16* __restrict__ AttT,       // [B][m][n]
    int N, int C4) {
  extern __shared__ char smem_raw[];
  const int LDE = N + 4;
  float* e      = (float*)smem_raw;                                    // 16*LDE
  int*   mcol   = (int*)(smem_raw + (size_t)16 * LDE * sizeof(float)); // N
  int*   rowm   = mcol + N;                                            // 16
  float* rowmax = (float*)(rowm + 16);                                 // 16
  float* rowsum = rowmax + 16;                                         // 16
  __shared__ int   wsum[8];
  __shared__ float s_invcnt;

  const int b    = blockIdx.y;
  const int n0   = blockIdx.x * 16;
  const int tid  = threadIdx.x;
  const int wv_i = tid >> 5;
  const int lane = tid & 31;

  int csum = 0;
  for (int m = tid; m < N; m += 256) {
    int mm = mask[(size_t)b * N + m];
    mcol[m] = mm;
    csum += mm;
  }
  if (tid < 16) rowm[tid] = mask[(size_t)b * N + n0 + tid];
#pragma unroll
  for (int off = 16; off; off >>= 1) csum += __shfl_xor(csum, off, 32);
  if (lane == 0) wsum[wv_i] = csum;
  __syncthreads();
  if (tid == 0) {
    int c = 0;
#pragma unroll
    for (int i = 0; i < 8; ++i) c += wsum[i];
    s_invcnt = 1.0f / ((float)c + 1e-9f);
  }

  const _Float16* Aq = XQ + ((size_t)b * N + n0) * C4;
  v16h a0 = load_a_frag(Aq + 0,  C4);
  v16h a1 = load_a_frag(Aq + 32, C4);
  v16h a2 = load_a_frag(Aq + 64, C4);
  v16h a3 = load_a_frag(Aq + 96, C4);
  const int col = lane & 15;
  const int r0  = (lane >> 4) << 3;
  for (int t = wv_i * 16; t < wv_i * 16 + 16; ++t) {
    const _Float16* Bb = XKT + ((size_t)b * N + (size_t)t * 16) * C4;
    v8f acc = {};
    acc = wmma_f16(a0, load_b_frag(Bb + 0,  C4), acc);
    acc = wmma_f16(a1, load_b_frag(Bb + 32, C4), acc);
    acc = wmma_f16(a2, load_b_frag(Bb + 64, C4), acc);
    acc = wmma_f16(a3, load_b_frag(Bb + 96, C4), acc);
#pragma unroll
    for (int v = 0; v < 8; ++v)
      e[(size_t)(r0 + v) * LDE + t * 16 + col] = acc[v];
  }
  __syncthreads();

  const float invcnt = s_invcnt;
#pragma unroll
  for (int rr = 0; rr < 2; ++rr) {
    const int r  = wv_i * 2 + rr;
    const int mr = rowm[r];
    float mx = -3.4e38f;
    for (int m = lane; m < N; m += 32) {
      float t = att_xform(e[(size_t)r * LDE + m], mr, mcol[m], invcnt);
      e[(size_t)r * LDE + m] = t;
      mx = fmaxf(mx, t);
    }
#pragma unroll
    for (int off = 16; off; off >>= 1) mx = fmaxf(mx, __shfl_xor(mx, off, 32));
    float sm = 0.f;
    for (int m = lane; m < N; m += 32)
      sm += __expf(e[(size_t)r * LDE + m] - mx);
#pragma unroll
    for (int off = 16; off; off >>= 1) sm += __shfl_xor(sm, off, 32);
    if (lane == 0) { rowmax[r] = mx; rowsum[r] = 1.0f / sm; }
  }
  __syncthreads();

  for (int m = tid; m < N; m += 256) {
    v16h av;
#pragma unroll
    for (int i = 0; i < 16; ++i) {
      float t = e[(size_t)i * LDE + m];
      av[i] = (_Float16)(__expf(t - rowmax[i]) * rowsum[i]);
    }
    *(v16h*)(AttT + ((size_t)b * N + m) * N + n0) = av;
  }
}

// ---------------------------------------------------------------------------
static inline char* carve(char*& p, size_t bytes) {
  char* r = p;
  p += (bytes + 255) & ~(size_t)255;
  return r;
}

extern "C" void kernel_launch(void* const* d_in, const int* in_sizes, int n_in,
                              void* d_out, int out_size, void* d_ws, size_t ws_size,
                              hipStream_t stream) {
  (void)in_sizes; (void)n_in; (void)out_size; (void)ws_size;
  const float* x     = (const float*)d_in[0];
  const float* query = (const float*)d_in[1];
  const int*   mask  = (const int*)d_in[2];
  const float* wqk   = (const float*)d_in[3];
  const float* wv    = (const float*)d_in[4];
  const float* bv    = (const float*)d_in[5];
  const float* wt    = (const float*)d_in[6];
  const float* bt    = (const float*)d_in[7];
  const float* gamma = (const float*)d_in[8];
  const float* beta  = (const float*)d_in[9];
  const float* rmean = (const float*)d_in[10];
  const float* rvar  = (const float*)d_in[11];
  float* out = (float*)d_out;

  const int B = AB_, C = AC_, N = AN_, C4 = AC4_;

  char* p = (char*)d_ws;
  _Float16* wqk_h = (_Float16*)carve(p, (size_t)C4 * C * 2);
  _Float16* wv_h  = (_Float16*)carve(p, (size_t)C * C * 2);
  _Float16* wt_h  = (_Float16*)carve(p, (size_t)C * C * 2);
  _Float16* xT_h  = (_Float16*)carve(p, (size_t)B * N * C * 2);
  _Float16* qT_h  = (_Float16*)carve(p, (size_t)B * N * C * 2);
  _Float16* xq_h  = (_Float16*)carve(p, (size_t)B * N * C4 * 2);
  _Float16* xkT_h = (_Float16*)carve(p, (size_t)B * N * C4 * 2);
  _Float16* xv_h  = (_Float16*)carve(p, (size_t)B * C * N * 2);
  _Float16* att_h = (_Float16*)carve(p, (size_t)B * N * N * 2);
  _Float16* hT_h  = (_Float16*)carve(p, (size_t)B * N * C * 2);

  // 1) weights -> f16
  k_f32_to_f16<<<(C4 * C) / 256, 256, 0, stream>>>(wqk, wqk_h, C4 * C);
  k_f32_to_f16<<<(C * C) / 256, 256, 0, stream>>>(wv, wv_h, C * C);
  k_f32_to_f16<<<(C * C) / 256, 256, 0, stream>>>(wt, wt_h, C * C);

  // 2) transpose + convert activations
  dim3 tg(N / 32, C / 32, B), tb(32, 8);
  k_transpose_f16<<<tg, tb, 0, stream>>>(x, xT_h, C, N);
  k_transpose_f16<<<tg, tb, 0, stream>>>(query, qT_h, C, N);

  // 3) x_q = qT @ wqk^T : [B][N][C4]; x_kT likewise from xT  (BN = 128)
  {
    int Mt = N / 64, Nt = C4 / 128;  // 32 x 1
    int blocks = B * Mt * Nt;
    k_gemm_big<0, 2><<<blocks, 256, 0, stream>>>(qT_h, (size_t)N * C, C,
                                                 wqk_h, 0, C,
                                                 xq_h, (size_t)N * C4, C4,
                                                 nullptr, nullptr, 0, nullptr, nullptr,
                                                 nullptr, nullptr, nullptr, nullptr,
                                                 nullptr, Mt, Nt, C, N, C);
    k_gemm_big<0, 2><<<blocks, 256, 0, stream>>>(xT_h, (size_t)N * C, C,
                                                 wqk_h, 0, C,
                                                 xkT_h, (size_t)N * C4, C4,
                                                 nullptr, nullptr, 0, nullptr, nullptr,
                                                 nullptr, nullptr, nullptr, nullptr,
                                                 nullptr, Mt, Nt, C, N, C);
  }

  // 4) x_v = wv @ x + bv : [B][C][N] f16  (BN = 256)
  {
    int Mt = C / 64, Nt = N / 256;   // 8 x 8
    int blocks = B * Mt * Nt;
    k_gemm_big<1, 4><<<blocks, 256, 0, stream>>>(wv_h, 0, C,
                                                 xT_h, (size_t)N * C, C,
                                                 xv_h, (size_t)C * N, N,
                                                 bv, nullptr, 0, nullptr, nullptr,
                                                 nullptr, nullptr, nullptr, nullptr,
                                                 nullptr, Mt, Nt, C, N, C);
  }

  // 5) masked-softmax attention, written transposed [B][m][n]
  {
    size_t smem = (size_t)16 * (N + 4) * sizeof(float) + (size_t)N * sizeof(int)
                + 16 * sizeof(int) + 32 * sizeof(float);
    k_attention<<<dim3(N / 16, B), 256, smem, stream>>>(xq_h, xkT_h, mask, att_h, N, C4);
  }

  // 6) hT = (x - x_v @ att)^T : [B][N][C] f16  (K = 2048)
  {
    int Mt = C / 64, Nt = N / 256;
    int blocks = B * Mt * Nt;
    k_gemm_big<2, 4><<<blocks, 256, 0, stream>>>(xv_h, (size_t)C * N, N,
                                                 att_h, (size_t)N * N, N,
                                                 hT_h, (size_t)N * C, 0,
                                                 nullptr, xT_h, (size_t)N * C,
                                                 nullptr, nullptr, nullptr, nullptr,
                                                 nullptr, nullptr, nullptr,
                                                 Mt, Nt, N, N, C);
  }

  // 7) out = x + relu(BN(wt @ h + bt))
  {
    int Mt = C / 64, Nt = N / 256;
    int blocks = B * Mt * Nt;
    k_gemm_big<3, 4><<<blocks, 256, 0, stream>>>(wt_h, 0, C,
                                                 hT_h, (size_t)N * C, C,
                                                 nullptr, 0, 0,
                                                 nullptr, nullptr, 0,
                                                 x, bt, gamma, beta, rmean, rvar,
                                                 out, Mt, Nt, C, N, C);
  }
}
